// TinyLoRALinear_38697655337004
// MI455X (gfx1250) — compile-verified, matrix operand order checked by reading, not run on hardware
//
#include <hip/hip_runtime.h>

// ---- CDNA5 (gfx1250) types -------------------------------------------------
typedef __attribute__((ext_vector_type(16))) __bf16 v16bf; // 8 VGPRs of bf16x2
typedef __attribute__((ext_vector_type(8)))  float  v8f;   // 16x16 f32 C/D tile
typedef __attribute__((ext_vector_type(4)))  float  f32x4;
typedef __attribute__((ext_vector_type(2)))  float  f32x2;

__device__ __forceinline__ v8f wmma_bf16(v16bf a, v16bf b, v8f c) {
    // D = A(16x32 bf16) x B(32x16 bf16) + C(16x16 f32)
    return __builtin_amdgcn_wmma_f32_16x16x32_bf16(
        /*neg_a=*/false, a, /*neg_b=*/false, b,
        /*c_mod=*/(short)0, c, /*reuse_a=*/false, /*reuse_b=*/false);
}

// Problem constants (from reference): D_in = D_out = 2048, R = 2, U_dim = 16
#define KDIM 2048
#define NDIM 2048
#define BM   128
#define BN   128
#define BK   32
#define NKT  (KDIM / BK)   // 64 k-tiles
#define LDSS 40            // padded LDS row stride in bf16 elems (80B, conflict-free)
#define BUFE (BM * LDSS)   // elems per single-buffer tile

__global__ __launch_bounds__(256)
void tinylora_wmma_gemm(const float* __restrict__ x,
                        const float* __restrict__ W,
                        const float* __restrict__ bias,
                        const float* __restrict__ U,
                        const float* __restrict__ S,
                        const float* __restrict__ Vh,
                        const float* __restrict__ P,
                        const float* __restrict__ gv,
                        float* __restrict__ out) {
    // Double-buffered bf16 hi/lo tiles: 2 * 4 * 10240 B = 80 KB (WGP has 320 KB)
    __shared__ __bf16 sAhi[2 * BUFE];
    __shared__ __bf16 sAlo[2 * BUFE];
    __shared__ __bf16 sBhi[2 * BUFE];
    __shared__ __bf16 sBlo[2 * BUFE];

    const int tid   = threadIdx.x;
    const int wave  = tid >> 5;
    const int lane  = tid & 31;
    const int wm    = wave & 3;      // 4 wave-rows  -> 32 M-rows each
    const int wn    = wave >> 2;     // 2 wave-cols  -> 64 N-cols each
    const int lrow  = lane & 15;     // M (or N) within 16x16 tile
    const int lhalf = lane >> 4;     // K half-select per ISA A/B layout

    const int blockM = blockIdx.y * BM;
    const int blockN = blockIdx.x * BN;

    // Loader slot mapping: slot i covers row0+32*i, same c4 for all i
    const int row0 = tid >> 3;          // 0..31
    const int c4   = (tid & 7) << 2;    // k offset within tile (same for all slots)

    // LoRA scale: svp[r] = S[r] * sum_u gv[u] * P[u][r]   (rank 2, u_dim 16)
    float svp0 = 0.f, svp1 = 0.f;
#pragma unroll
    for (int u = 0; u < 16; ++u) {
        const float g = gv[u];
        svp0 += g * P[2 * u + 0];
        svp1 += g * P[2 * u + 1];
    }
    svp0 *= S[0];
    svp1 *= S[1];

    // k-invariant per-slot U coefficients (hoisted out of the k-loop)
    f32x2 ur[4];
#pragma unroll
    for (int i = 0; i < 4; ++i)
        ur[i] = *(const f32x2*)(U + 2 * (blockN + row0 + 32 * i));

    const float* xbase  = x  + (size_t)(blockM + row0) * KDIM + c4;
    const float* wbase  = W  + (size_t)(blockN + row0) * KDIM + c4;
    const float* vbase0 = Vh + c4;
    const float* vbase1 = Vh + KDIM + c4;

    // Register staging for the in-flight tile (issued early, consumed late)
    f32x4 avr[4], wvr[4], v0r, v1r;

    auto load_regs = [&](int kt) {
#pragma unroll
        for (int i = 0; i < 4; ++i) {
            avr[i] = *(const f32x4*)(xbase + (size_t)i * 32 * KDIM + kt);
            wvr[i] = *(const f32x4*)(wbase + (size_t)i * 32 * KDIM + kt);
        }
        v0r = *(const f32x4*)(vbase0 + kt);
        v1r = *(const f32x4*)(vbase1 + kt);
        // pull the tile after next toward L2 (global_prefetch_b8)
        if (kt + BK < KDIM) {
            __builtin_prefetch(xbase + kt + BK, 0, 1);
            __builtin_prefetch(wbase + kt + BK, 0, 1);
        }
    };

    auto store_lds = [&](int buf) {
        __bf16* pAhi = sAhi + buf * BUFE;
        __bf16* pAlo = sAlo + buf * BUFE;
        __bf16* pBhi = sBhi + buf * BUFE;
        __bf16* pBlo = sBlo + buf * BUFE;
#pragma unroll
        for (int i = 0; i < 4; ++i) {
            const int r = row0 + 32 * i;
            // exact f32 LoRA fold before bf16 split
            f32x4 wv = wvr[i] + (svp0 * ur[i][0]) * v0r + (svp1 * ur[i][1]) * v1r;
#pragma unroll
            for (int e = 0; e < 4; ++e) {
                const float  af = avr[i][e];
                const __bf16 ah = (__bf16)af;
                pAhi[r * LDSS + c4 + e] = ah;
                pAlo[r * LDSS + c4 + e] = (__bf16)(af - (float)ah);
                const float  wf = wv[e];
                const __bf16 wh = (__bf16)wf;
                pBhi[r * LDSS + c4 + e] = wh;
                pBlo[r * LDSS + c4 + e] = (__bf16)(wf - (float)wh);
            }
        }
    };

    v8f acc[2][4] = {};  // 2 M-tiles x 4 N-tiles of 16x16 f32 per wave

    auto compute = [&](int buf) {
        const __bf16* pAhi = sAhi + buf * BUFE;
        const __bf16* pAlo = sAlo + buf * BUFE;
        const __bf16* pBhi = sBhi + buf * BUFE;
        const __bf16* pBlo = sBlo + buf * BUFE;

        // A fragments (ISA 16-bit A layout: two b128 loads per fragment)
        v16bf ah[2], al[2];
#pragma unroll
        for (int mi = 0; mi < 2; ++mi) {
            const int r = wm * 32 + mi * 16 + lrow;
            const __bf16* ph = &pAhi[r * LDSS + lhalf * 8];
            ((f32x4*)&ah[mi])[0] = *(const f32x4*)(ph);       // K = half*8 .. +7
            ((f32x4*)&ah[mi])[1] = *(const f32x4*)(ph + 16);  // K = 16+half*8 .. +7
            const __bf16* pl = &pAlo[r * LDSS + lhalf * 8];
            ((f32x4*)&al[mi])[0] = *(const f32x4*)(pl);
            ((f32x4*)&al[mi])[1] = *(const f32x4*)(pl + 16);
        }

        // B fragments + 3-term split-accumulate WMMA
#pragma unroll
        for (int ni = 0; ni < 4; ++ni) {
            const int r = wn * 64 + ni * 16 + lrow;           // N column = W row
            v16bf bh, bl;
            const __bf16* ph = &pBhi[r * LDSS + lhalf * 8];
            ((f32x4*)&bh)[0] = *(const f32x4*)(ph);
            ((f32x4*)&bh)[1] = *(const f32x4*)(ph + 16);
            const __bf16* pl = &pBlo[r * LDSS + lhalf * 8];
            ((f32x4*)&bl)[0] = *(const f32x4*)(pl);
            ((f32x4*)&bl)[1] = *(const f32x4*)(pl + 16);

#pragma unroll
            for (int mi = 0; mi < 2; ++mi) {
                acc[mi][ni] = wmma_bf16(ah[mi], bh, acc[mi][ni]); // hi*hi
                acc[mi][ni] = wmma_bf16(ah[mi], bl, acc[mi][ni]); // hi*lo
                acc[mi][ni] = wmma_bf16(al[mi], bh, acc[mi][ni]); // lo*hi
            }
        }
    };

    // ---- software pipeline: prologue fills buffer 0 -------------------------
    load_regs(0);
    store_lds(0);

    int buf = 0;
#pragma unroll 1
    for (int t = 0; t < NKT; ++t) {
        __syncthreads();                       // lds[buf] ready for all waves
        if (t + 1 < NKT) load_regs((t + 1) * BK);  // next tile in flight
        compute(buf);                          // 24 WMMAs over lds[buf]
        if (t + 1 < NKT) store_lds(buf ^ 1);   // fill the other buffer
        buf ^= 1;
    }

    // ---- epilogue: bias add, store (C/D layout: N = lane%16, M = v + 8*(lane/16))
#pragma unroll
    for (int ni = 0; ni < 4; ++ni) {
        const int n  = blockN + wn * 64 + ni * 16 + lrow;
        const float bv = bias[n];
#pragma unroll
        for (int mi = 0; mi < 2; ++mi) {
            const int mbase = blockM + wm * 32 + mi * 16 + lhalf * 8;
#pragma unroll
            for (int v = 0; v < 8; ++v) {
                out[(size_t)(mbase + v) * NDIM + n] = acc[mi][ni][v] + bv;
            }
        }
    }
}

extern "C" void kernel_launch(void* const* d_in, const int* in_sizes, int n_in,
                              void* d_out, int out_size, void* d_ws, size_t ws_size,
                              hipStream_t stream) {
    const float* x    = (const float*)d_in[0];
    const float* Wb   = (const float*)d_in[1];
    const float* bias = (const float*)d_in[2];
    const float* U    = (const float*)d_in[3];
    const float* S    = (const float*)d_in[4];
    const float* Vh   = (const float*)d_in[5];
    const float* P    = (const float*)d_in[6];
    const float* gv   = (const float*)d_in[7];
    float* out = (float*)d_out;

    const int M = in_sizes[0] / KDIM;        // B * S_len = 16384
    dim3 grid(NDIM / BN, M / BM);            // (16, 128)
    dim3 block(256);                         // 8 wave32s
    tinylora_wmma_gemm<<<grid, block, 0, stream>>>(x, Wb, bias, U, S, Vh, P, gv, out);
}
